// MemoryModule_21397527069133
// MI455X (gfx1250) — matrix operand order repaired and checked by reading, not compiled, not run on hardware
//
#include <hip/hip_runtime.h>

#define REP_DIM   256
#define ATT_DIM   256
#define AGRU_DIM  256
#define BATCH     64
#define MAX_FACTS 512

// ---------------------------------------------------------------------------
// CDNA5 WMMA types (wave32): 16x16x32 bf16 -> f32
// ---------------------------------------------------------------------------
typedef __attribute__((ext_vector_type(16))) __bf16       v16bf;
typedef __attribute__((ext_vector_type(8)))  float        v8f;
typedef __attribute__((ext_vector_type(4)))  unsigned int v4u;

__device__ __forceinline__ unsigned short f2bf(float f) {
  unsigned int x = __float_as_uint(f);
  unsigned int r = ((x >> 16) & 1u) + 0x7fffu;   // round-to-nearest-even
  return (unsigned short)((x + r) >> 16);
}
__device__ __forceinline__ float bf2f(unsigned short u) {
  return __uint_as_float((unsigned)u << 16);
}
__device__ __forceinline__ float fast_tanh(float x) {
#if __has_builtin(__builtin_amdgcn_tanhf)
  return __builtin_amdgcn_tanhf(x);          // gfx1250 v_tanh_f32
#elif __has_builtin(__builtin_amdgcn_tanh_f32)
  return __builtin_amdgcn_tanh_f32(x);
#else
  return tanhf(x);
#endif
}
__device__ __forceinline__ float fast_sigmoid(float x) {
  return 1.f / (1.f + __expf(-x));           // v_exp_f32
}

// CDNA5 async global->LDS copies (ASYNCcnt-tracked, bypass VGPRs)
__device__ __forceinline__ void async_g2l_b128(void* lds_ptr, const void* gptr) {
  unsigned l = (unsigned)(unsigned long long)lds_ptr;   // low 32b = LDS offset
  asm volatile("global_load_async_to_lds_b128 %0, %1, off"
               :: "v"(l), "v"(gptr) : "memory");
}
__device__ __forceinline__ void async_g2l_b32(void* lds_ptr, const void* gptr) {
  unsigned l = (unsigned)(unsigned long long)lds_ptr;
  asm volatile("global_load_async_to_lds_b32 %0, %1, off"
               :: "v"(l), "v"(gptr) : "memory");
}
__device__ __forceinline__ void wait_async0() {
  asm volatile("s_wait_asynccnt 0x0" ::: "memory");
}

union BfFrag { v16bf v; v4u u[2]; };

// A-matrix (16x32 bf16) fragment. Lane holds row M=lane&15.
// lanes<16: e0..7->K=kb..kb+7, e8..15->K=kb+16..23 ; lanes>=16: +8 / +24
__device__ __forceinline__ v16bf load_afrag(const unsigned short* rowbase, int kb, int hi) {
  BfFrag f;
  f.u[0] = *(const v4u*)(rowbase + kb + hi * 8);
  f.u[1] = *(const v4u*)(rowbase + kb + 16 + hi * 8);
  return f.v;
}
// B-matrix (32x16 bf16, B[k][n]=W[n][k]). Lane holds col N=lane&15;
// e0..7->K=kb+hi*8.., e8..15->K=kb+16+hi*8.. (same order as A).
__device__ __forceinline__ v16bf load_bfrag(const unsigned short* rowbase, int kb, int hi) {
  BfFrag f;
  f.u[0] = *(const v4u*)(rowbase + kb + hi * 8);
  f.u[1] = *(const v4u*)(rowbase + kb + 16 + hi * 8);
  return f.v;
}
__device__ __forceinline__ v8f wmma_bf16(v16bf a, v16bf b, v8f c) {
  return __builtin_amdgcn_wmma_f32_16x16x32_bf16(false, a, false, b, (short)0, c,
                                                 false, false);
}

// ---------------------------------------------------------------------------
// Kernel 0: f32 -> bf16(raw u16) conversion into workspace
// ---------------------------------------------------------------------------
__global__ __launch_bounds__(256) void cvt_f32_bf16(const float* __restrict__ src,
                                                    unsigned short* __restrict__ dst,
                                                    int n) {
  for (int i = blockIdx.x * blockDim.x + threadIdx.x; i < n;
       i += gridDim.x * blockDim.x)
    dst[i] = f2bf(src[i]);
}

// ---------------------------------------------------------------------------
// Kernel 1: X_r = facts @ Wr^T, X_g = facts @ Wg^T  (bf16 out).
// Fully parallel: one block per (batch, 16-fact tile). A-fragments hoisted
// and reused across 4 output tiles per wave (2 matrices x 2 N-tiles).
// ---------------------------------------------------------------------------
__global__ __launch_bounds__(256) void xgemm_kernel(
    const unsigned short* __restrict__ factsbf,
    const unsigned short* __restrict__ wrbf, const unsigned short* __restrict__ wgbf,
    unsigned short* __restrict__ xr, unsigned short* __restrict__ xg) {
  const int b = blockIdx.y, t0 = blockIdx.x * 16;
  const int tid = threadIdx.x, wave = tid >> 5, lane = tid & 31;
  const int hi = lane >> 4, lc = lane & 15;
  const unsigned short* arow =
      factsbf + ((size_t)b * MAX_FACTS + t0 + lc) * REP_DIM;
  v16bf af[8];
#pragma unroll
  for (int kc = 0; kc < 8; ++kc) af[kc] = load_afrag(arow, kc * 32, hi);
#pragma unroll
  for (int mt = 0; mt < 2; ++mt) {
    const unsigned short* wsrc = mt ? wgbf : wrbf;
    unsigned short* xout = mt ? xg : xr;
#pragma unroll
    for (int j = 0; j < 2; ++j) {
      const int n = (wave + j * 8) * 16 + lc;
      const unsigned short* brow = wsrc + (size_t)n * AGRU_DIM;
      v8f acc = {};
#pragma unroll
      for (int kc = 0; kc < 8; ++kc)
        acc = wmma_bf16(af[kc], load_bfrag(brow, kc * 32, hi), acc);
#pragma unroll
      for (int r = 0; r < 8; ++r) {
        int M = r + hi * 8;
        xout[((size_t)b * MAX_FACTS + t0 + M) * AGRU_DIM + n] = f2bf(acc[r]);
      }
    }
  }
}

// ---------------------------------------------------------------------------
// Kernel 2: attention scores. Facts tile async-staged to LDS, z built in bf16
// LDS, Z = tanh(z@W1^T + b1) via WMMA (2 accumulators share each A-fragment),
// score = Z@W2^T + b2 with wave32 half-lane reductions.
// ---------------------------------------------------------------------------
__global__ __launch_bounds__(256) void attn_scores_kernel(
    const float* __restrict__ queries, const float* __restrict__ facts,
    const float* __restrict__ m_old, const unsigned short* __restrict__ w1bf,
    const float* __restrict__ w1b, const float* __restrict__ w2w,
    const float* __restrict__ w2b, float* __restrict__ scores) {
  __shared__ float q_s[REP_DIM];
  __shared__ float m_s[REP_DIM];
  __shared__ alignas(16) float f_s[16 * REP_DIM];
  __shared__ alignas(16) unsigned short z_s[16 * 1024];
  __shared__ float wave_part[8][16];

  const int b   = blockIdx.y;
  const int t0  = blockIdx.x * 16;
  const int tid = threadIdx.x;

  // async-stage the 16x256 f32 facts tile (16KB): 1024 16B chunks
#pragma unroll
  for (int k = 0; k < 4; ++k) {
    int c = tid + k * 256;
    int row = c >> 6, off = (c & 63) * 4;
    async_g2l_b128(f_s + row * REP_DIM + off,
                   facts + ((size_t)b * MAX_FACTS + t0 + row) * REP_DIM + off);
  }
  q_s[tid] = queries[b * REP_DIM + tid];
  m_s[tid] = m_old[b * REP_DIM + tid];
  wait_async0();
  __syncthreads();

  for (int i = tid; i < 16 * 1024; i += 256) {
    int row = i >> 10;
    int f   = i & 1023;
    int k   = f & 255;
    int sel = f >> 8;
    float fv = f_s[row * REP_DIM + k];
    float qv = q_s[k], mv = m_s[k];
    float val = (sel == 0)   ? fv * qv
                : (sel == 1) ? fv * mv
                : (sel == 2) ? fabsf(fv - qv)
                             : fabsf(fv - mv);
    z_s[i] = f2bf(val);
  }
  __syncthreads();

  const int wave = tid >> 5, lane = tid & 31;
  const int hi = lane >> 4, lc = lane & 15;
  const unsigned short* arow  = z_s + lc * 1024;
  const unsigned short* brow0 = w1bf + (size_t)(wave * 16 + lc) * 1024;
  const unsigned short* brow1 = w1bf + (size_t)((wave + 8) * 16 + lc) * 1024;

  v8f acc0 = {}, acc1 = {};
#pragma unroll 4
  for (int kc = 0; kc < 32; ++kc) {
    v16bf a = load_afrag(arow, kc * 32, hi);
    acc0 = wmma_bf16(a, load_bfrag(brow0, kc * 32, hi), acc0);
    acc1 = wmma_bf16(a, load_bfrag(brow1, kc * 32, hi), acc1);
  }
  float b0 = w1b[wave * 16 + lc],       w20 = w2w[wave * 16 + lc];
  float b1 = w1b[(wave + 8) * 16 + lc], w21 = w2w[(wave + 8) * 16 + lc];
  float p[8];
#pragma unroll
  for (int r = 0; r < 8; ++r)
    p[r] = fast_tanh(acc0[r] + b0) * w20 + fast_tanh(acc1[r] + b1) * w21;

#pragma unroll
  for (int msk = 8; msk >= 1; msk >>= 1)
#pragma unroll
    for (int r = 0; r < 8; ++r) p[r] += __shfl_xor(p[r], msk, 32);
  if (lc == 0)
#pragma unroll
    for (int r = 0; r < 8; ++r) wave_part[wave][hi * 8 + r] = p[r];
  __syncthreads();
  if (tid < 16) {
    float s = w2b[0];
    for (int w = 0; w < 8; ++w) s += wave_part[w][tid];
    scores[b * MAX_FACTS + t0 + tid] = s;
  }
}

// ---------------------------------------------------------------------------
// Kernel 3: masked softmax over T per batch
// ---------------------------------------------------------------------------
__global__ __launch_bounds__(256) void softmax_kernel(const float* __restrict__ scores,
                                                      const int* __restrict__ num_facts,
                                                      float* __restrict__ g) {
  __shared__ float red[256];
  const int b = blockIdx.x, tid = threadIdx.x;
  const int nf = num_facts[b];
  float v0 = (tid < nf) ? scores[b * MAX_FACTS + tid] : -INFINITY;
  float v1 = (tid + 256 < nf) ? scores[b * MAX_FACTS + tid + 256] : -INFINITY;
  red[tid] = fmaxf(v0, v1);
  __syncthreads();
  for (int s = 128; s > 0; s >>= 1) {
    if (tid < s) red[tid] = fmaxf(red[tid], red[tid + s]);
    __syncthreads();
  }
  float mx = red[0];
  __syncthreads();
  float e0 = (tid < nf) ? __expf(v0 - mx) : 0.f;
  float e1 = (tid + 256 < nf) ? __expf(v1 - mx) : 0.f;
  red[tid] = e0 + e1;
  __syncthreads();
  for (int s = 128; s > 0; s >>= 1) {
    if (tid < s) red[tid] += red[tid + s];
    __syncthreads();
  }
  float inv = 1.f / red[0];
  g[b * MAX_FACTS + tid]       = e0 * inv;
  g[b * MAX_FACTS + tid + 256] = e1 * inv;
}

// ---------------------------------------------------------------------------
// Kernel 4: AGRU scan. One block = 16 batches (WMMA M-dim).
// LDS (~304KB of CDNA5's 320KB WGP LDS): Ur+Ug bf16 resident (256KB, async-
// loaded), double-buffered bf16 h (16KB), double-buffered X_r/X_g step tiles
// (32KB) and g_t scalars -- all prefetched one step ahead with
// global_load_async_to_lds_* while the current step computes.
// Per step only h@Ur^T, h@Ug^T remain (WMMA), each split into two
// independent 4-deep accumulator chains. h_old is register-resident (each
// lane owns its (M,n) cells), so a single barrier per step suffices.
// ---------------------------------------------------------------------------
__global__ __launch_bounds__(256) void agru_scan_kernel(
    const unsigned short* __restrict__ xr, const unsigned short* __restrict__ xg,
    const unsigned short* __restrict__ urbf, const unsigned short* __restrict__ ugbf,
    const float* __restrict__ urb, const float* __restrict__ ugb,
    const float* __restrict__ m_old, const float* __restrict__ gmat,
    const int* __restrict__ num_facts, float* __restrict__ cbuf) {
  extern __shared__ char smem[];
  unsigned short* ur_s = (unsigned short*)smem;               // 256x256 bf16
  unsigned short* ug_s = ur_s + AGRU_DIM * AGRU_DIM;          // 256x256 bf16
  unsigned short* h_bf = ug_s + AGRU_DIM * AGRU_DIM;          // 2 x 16x256 bf16
  unsigned short* xr_b = h_bf + 2 * 4096;                     // 2 x 16x256 bf16
  unsigned short* xg_b = xr_b + 2 * 4096;                     // 2 x 16x256 bf16
  float*          g_s  = (float*)(xg_b + 2 * 4096);           // 2 x 16 f32

  const int tid = threadIdx.x;
  const int b0  = blockIdx.x * 16;

  // async copy Ur,Ug (128KB each) into LDS: 8192 16B chunks each
  for (int c = tid; c < 8192; c += 256) {
    async_g2l_b128(ur_s + c * 8, urbf + c * 8);
    async_g2l_b128(ug_s + c * 8, ugbf + c * 8);
  }
  // h(0) = m_old (bf16, buffer 0)
  for (int i = tid; i < 16 * 256; i += 256) {
    int row = i >> 8, n = i & 255;
    h_bf[i] = f2bf(m_old[(b0 + row) * REP_DIM + n]);
  }
  // preload X tiles and g scalars for t=0 into buffer 0
  for (int c = tid; c < 512; c += 256) {
    int row = c >> 5, off = (c & 31) * 8;
    size_t src = ((size_t)(b0 + row) * MAX_FACTS) * AGRU_DIM + off;
    async_g2l_b128(xr_b + row * 256 + off, xr + src);
    async_g2l_b128(xg_b + row * 256 + off, xg + src);
  }
  if (tid < 16)
    async_g2l_b32(g_s + tid, gmat + (size_t)(b0 + tid) * MAX_FACTS);
  wait_async0();
  __syncthreads();

  const int wave = tid >> 5, lane = tid & 31;
  const int hi = lane >> 4, lc = lane & 15;
  const int n0 = wave * 16 + lc, n1 = (wave + 8) * 16 + lc;

  int nfr[8];
#pragma unroll
  for (int r = 0; r < 8; ++r) nfr[r] = num_facts[b0 + r + hi * 8];
  const float br0 = urb[n0], bg0 = ugb[n0], br1 = urb[n1], bg1 = ugb[n1];

  // register-resident h_old for the cells this lane owns
  float hold[2][8];
#pragma unroll
  for (int q2 = 0; q2 < 2; ++q2)
#pragma unroll
    for (int r = 0; r < 8; ++r)
      hold[q2][r] = m_old[(b0 + r + hi * 8) * REP_DIM + (q2 ? n1 : n0)];

  for (int t = 0; t < MAX_FACTS; ++t) {
    const int cur = t & 1, nxt = cur ^ 1;
    // prefetch next step's X tiles + g scalars into the other LDS buffers
    if (t + 1 < MAX_FACTS) {
      for (int c = tid; c < 512; c += 256) {
        int row = c >> 5, off = (c & 31) * 8;
        size_t src = ((size_t)(b0 + row) * MAX_FACTS + (t + 1)) * AGRU_DIM + off;
        async_g2l_b128(xr_b + nxt * 4096 + row * 256 + off, xr + src);
        async_g2l_b128(xg_b + nxt * 4096 + row * 256 + off, xg + src);
      }
      if (tid < 16)
        async_g2l_b32(g_s + nxt * 16 + tid,
                      gmat + (size_t)(b0 + tid) * MAX_FACTS + (t + 1));
    }

    const unsigned short* h_row = h_bf + cur * 4096 + lc * 256;
    float gt[8];
#pragma unroll
    for (int r = 0; r < 8; ++r) gt[r] = g_s[cur * 16 + r + hi * 8];

    float hnew[2][8];
#pragma unroll
    for (int q2 = 0; q2 < 2; ++q2) {
      const int n = q2 ? n1 : n0;
      const unsigned short* bu_r = ur_s + n * AGRU_DIM;
      const unsigned short* bu_g = ug_s + n * AGRU_DIM;
      // hr0 starts from X_r (r-gate pre-activation); X_g kept separate for
      // the r-gated update:  h_t = tanh(Xg + r*(h@Ug + bg))
      v8f hr0, hr1 = {}, hg0 = {}, hg1 = {};
      float xgv[8];
#pragma unroll
      for (int r = 0; r < 8; ++r) {
        int M = r + hi * 8;
        hr0[r] = bf2f(xr_b[cur * 4096 + M * 256 + n]);
        xgv[r] = bf2f(xg_b[cur * 4096 + M * 256 + n]);
      }
      // two independent 4-deep WMMA chains (even/odd K-chunks)
#pragma unroll
      for (int kc = 0; kc < 8; kc += 2) {
        int kbe = kc * 32, kbo = (kc + 1) * 32;
        v16bf ahe = load_afrag(h_row, kbe, hi);
        v16bf aho = load_afrag(h_row, kbo, hi);
        hr0 = wmma_bf16(ahe, load_bfrag(bu_r, kbe, hi), hr0);
        hg0 = wmma_bf16(ahe, load_bfrag(bu_g, kbe, hi), hg0);
        hr1 = wmma_bf16(aho, load_bfrag(bu_r, kbo, hi), hr1);
        hg1 = wmma_bf16(aho, load_bfrag(bu_g, kbo, hi), hg1);
      }
      v8f ar = hr0 + hr1;   // X_r + h@Ur
      v8f ag = hg0 + hg1;   // h@Ug
      const float br = q2 ? br1 : br0, bg = q2 ? bg1 : bg0;
#pragma unroll
      for (int r = 0; r < 8; ++r) {
        float rr  = fast_sigmoid(ar[r] + br);
        float htv = fast_tanh(xgv[r] + rr * (ag[r] + bg));
        hnew[q2][r] = gt[r] * htv + (1.f - gt[r]) * hold[q2][r];
      }
    }

    // write h(t+1) into the other buffer; emit c rows at t == nf-1
#pragma unroll
    for (int q2 = 0; q2 < 2; ++q2) {
      const int n = q2 ? n1 : n0;
#pragma unroll
      for (int r = 0; r < 8; ++r) {
        int M = r + hi * 8;
        float v = hnew[q2][r];
        h_bf[nxt * 4096 + M * 256 + n] = f2bf(v);
        hold[q2][r] = v;
        if (t == nfr[r] - 1) cbuf[(size_t)(b0 + M) * REP_DIM + n] = v;
      }
    }
    wait_async0();     // our async X/g(t+1) writes have landed
    __syncthreads();   // single barrier: h(t+1), X(t+1), g(t+1) all visible
  }
}

// ---------------------------------------------------------------------------
// Kernel 5: m_new = relu(concat(m_old, c, q) @ Wt^T + b). Tiny.
// ---------------------------------------------------------------------------
__global__ __launch_bounds__(256) void final_kernel(
    const float* __restrict__ m_old, const float* __restrict__ cbuf,
    const float* __restrict__ queries, const float* __restrict__ wt,
    const float* __restrict__ wtb, float* __restrict__ out) {
  const int b = blockIdx.x, o = threadIdx.x;
  const float* wrow = wt + (size_t)o * (2 * REP_DIM + AGRU_DIM);
  float s = wtb[o];
  for (int j = 0; j < 256; ++j) s += m_old[b * 256 + j] * wrow[j];
  for (int j = 0; j < 256; ++j) s += cbuf[b * 256 + j] * wrow[256 + j];
  for (int j = 0; j < 256; ++j) s += queries[b * 256 + j] * wrow[512 + j];
  out[b * 256 + o] = fmaxf(s, 0.f);
}

// ---------------------------------------------------------------------------
extern "C" void kernel_launch(void* const* d_in, const int* in_sizes, int n_in,
                              void* d_out, int out_size, void* d_ws, size_t ws_size,
                              hipStream_t stream) {
  (void)in_sizes; (void)n_in; (void)out_size; (void)ws_size;

  const float* queries   = (const float*)d_in[0];
  const float* facts     = (const float*)d_in[1];
  const int*   num_facts = (const int*)d_in[2];
  const float* m_old     = (const float*)d_in[3];
  const float* W1_w      = (const float*)d_in[4];
  const float* W1_b      = (const float*)d_in[5];
  const float* W2_w      = (const float*)d_in[6];
  const float* W2_b      = (const float*)d_in[7];
  const float* Wr_w      = (const float*)d_in[8];
  const float* Ur_w      = (const float*)d_in[9];
  const float* Ur_b      = (const float*)d_in[10];
  const float* Wg_w      = (const float*)d_in[11];
  const float* Ug_w      = (const float*)d_in[12];
  const float* Ug_b      = (const float*)d_in[13];
  const float* Wt_w      = (const float*)d_in[14];
  const float* Wt_b      = (const float*)d_in[15];
  float* out = (float*)d_out;

  char* ws = (char*)d_ws;
  size_t off = 0;
  auto take = [&](size_t bytes) -> char* {
    char* p = ws + off;
    off = (off + bytes + 255) & ~(size_t)255;
    return p;
  };
  unsigned short* w1bf    = (unsigned short*)take((size_t)ATT_DIM * 4 * REP_DIM * 2);
  unsigned short* wrbf    = (unsigned short*)take((size_t)AGRU_DIM * REP_DIM * 2);
  unsigned short* wgbf    = (unsigned short*)take((size_t)AGRU_DIM * REP_DIM * 2);
  unsigned short* urbf    = (unsigned short*)take((size_t)AGRU_DIM * AGRU_DIM * 2);
  unsigned short* ugbf    = (unsigned short*)take((size_t)AGRU_DIM * AGRU_DIM * 2);
  unsigned short* factsbf = (unsigned short*)take((size_t)BATCH * MAX_FACTS * REP_DIM * 2);
  unsigned short* xr      = (unsigned short*)take((size_t)BATCH * MAX_FACTS * AGRU_DIM * 2);
  unsigned short* xg      = (unsigned short*)take((size_t)BATCH * MAX_FACTS * AGRU_DIM * 2);
  float* scores = (float*)take((size_t)BATCH * MAX_FACTS * 4);
  float* gmat   = (float*)take((size_t)BATCH * MAX_FACTS * 4);
  float* cbuf   = (float*)take((size_t)BATCH * AGRU_DIM * 4);

  // 0) weight / fact bf16 conversion
  cvt_f32_bf16<<<256, 256, 0, stream>>>(W1_w, w1bf, ATT_DIM * 4 * REP_DIM);
  cvt_f32_bf16<<<64, 256, 0, stream>>>(Wr_w, wrbf, AGRU_DIM * REP_DIM);
  cvt_f32_bf16<<<64, 256, 0, stream>>>(Wg_w, wgbf, AGRU_DIM * REP_DIM);
  cvt_f32_bf16<<<64, 256, 0, stream>>>(Ur_w, urbf, AGRU_DIM * AGRU_DIM);
  cvt_f32_bf16<<<64, 256, 0, stream>>>(Ug_w, ugbf, AGRU_DIM * AGRU_DIM);
  cvt_f32_bf16<<<1024, 256, 0, stream>>>(facts, factsbf, BATCH * MAX_FACTS * REP_DIM);

  // 1) hoisted fact projections X_r, X_g (WMMA, fully parallel)
  dim3 xgrid(MAX_FACTS / 16, BATCH);
  xgemm_kernel<<<xgrid, 256, 0, stream>>>(factsbf, wrbf, wgbf, xr, xg);

  // 2) attention scores (WMMA + async LDS staging)
  dim3 agrid(MAX_FACTS / 16, BATCH);
  attn_scores_kernel<<<agrid, 256, 0, stream>>>(queries, facts, m_old, w1bf, W1_b,
                                                W2_w, W2_b, scores);

  // 3) masked softmax -> g
  softmax_kernel<<<BATCH, 256, 0, stream>>>(scores, num_facts, gmat);

  // 4) AGRU scan: ~304KB LDS (Ur/Ug resident + double-buffered async X/g/h)
  const size_t scan_lds = (size_t)AGRU_DIM * AGRU_DIM * 2 * 2  // Ur + Ug bf16
                        + (size_t)2 * 16 * 256 * 2             // h dbl-buf bf16
                        + (size_t)2 * 16 * 256 * 2 * 2         // X_r/X_g dbl-buf
                        + (size_t)2 * 16 * 4;                  // g_t dbl-buf
  agru_scan_kernel<<<BATCH / 16, 256, scan_lds, stream>>>(
      xr, xg, urbf, ugbf, Ur_b, Ug_b, m_old, gmat, num_facts, cbuf);

  // 5) output layer
  final_kernel<<<BATCH, 256, 0, stream>>>(m_old, cbuf, queries, Wt_w, Wt_b, out);
}
